// tiny_mixtral_4681514353107
// MI455X (gfx1250) — matrix operand order repaired and checked by reading, not compiled, not run on hardware
//
#include <hip/hip_runtime.h>
#include <hip/hip_bf16.h>
#include <cstddef>
#include <cstdint>

// ---------------- model constants ----------------
constexpr int S_ = 1024, V_ = 32000, D_ = 768;
constexpr int H_ = 12, DH_ = 64, KV_ = 4;
constexpr int E_ = 8, F_ = 2048, L_ = 2;
constexpr int WINDOW_ = 512;
constexpr int T_ = S_;                 // B*S
constexpr float EPS_ = 1e-5f;

typedef __attribute__((ext_vector_type(16))) __bf16 v16bf;
typedef __attribute__((ext_vector_type(8)))  float  v8f;

// ---------------- embedding gather ----------------
__global__ void embed_kernel(const int* __restrict__ x, const float* __restrict__ emb,
                             float* __restrict__ h) {
    int i = blockIdx.x * blockDim.x + threadIdx.x;
    if (i >= T_ * D_) return;
    int t = i / D_, d = i % D_;
    h[i] = emb[(size_t)x[t] * D_ + d];
}

// ---------------- RMSNorm (one block per token) ----------------
__global__ void rmsnorm_kernel(const float* __restrict__ x, const float* __restrict__ w,
                               float* __restrict__ out) {
    __shared__ float red[256];
    int t = blockIdx.x;
    const float* xr = x + (size_t)t * D_;
    float ss = 0.f;
    for (int d = threadIdx.x; d < D_; d += 256) { float v = xr[d]; ss += v * v; }
    red[threadIdx.x] = ss;
    __syncthreads();
    for (int s = 128; s > 0; s >>= 1) {
        if (threadIdx.x < s) red[threadIdx.x] += red[threadIdx.x + s];
        __syncthreads();
    }
    float r = rsqrtf(red[0] / (float)D_ + EPS_);
    float* orow = out + (size_t)t * D_;
    for (int d = threadIdx.x; d < D_; d += 256) orow[d] = xr[d] * w[d] * r;
}

// ---------------- generic bf16 WMMA GEMM, double-buffered async global->LDS ----------------
// C[M,N] = op( A[M,K] @ B[K,N] ), fp32 in/out, bf16 WMMA math.
// Tiles: BM=64, BN=64, BK=64 (two 16x16x32 WMMA K-steps per buffer).
// Software pipeline: GLOBAL_LOAD_ASYNC_TO_LDS_B128 (ASYNCcnt) streams tile i+1
// into the alternate LDS buffer while tile i is consumed by WMMA; partial
// s_wait_asynccnt 8 exploits in-order async completion. fp32->bf16 conversion
// is fused into the LDS->fragment gather (co-executes with WMMA).
__global__ __launch_bounds__(256)
void gemm_bf16_wmma(const float* __restrict__ A, const float* __restrict__ B,
                    float* __restrict__ C, const float* __restrict__ bias,
                    const float* __restrict__ rowscale, int rs_stride,
                    int M, int N, int K, int accum) {
    constexpr int BM = 64, BN = 64, BK = 64;
    constexpr int AP = 4, BP = 4;                 // row padding (floats) vs 64-bank LDS
    constexpr int ABUF = BM * (BK + AP);          // floats per A buffer
    constexpr int BBUF = BK * (BN + BP);          // floats per B buffer
    __shared__ float Asf[2][BM][BK + AP];
    __shared__ float Bsf[2][BK][BN + BP];

    const int tid  = threadIdx.x;
    const int wave = tid >> 5;          // 0..7
    const int lane = tid & 31;
    const int wm = wave & 3;            // 4 waves along M
    const int wn = wave >> 2;           // 2 waves along N
    const int mo = wm * 16;             // wave M offset in block
    const int no = wn * 32;             // wave N offset in block (2x 16-wide tiles)

    const int m0 = blockIdx.y * BM;
    const int n0 = blockIdx.x * BN;

    v8f acc0 = {}, acc1 = {};

    // cooperative async-load indices: 64 rows, 4 threads/row, 16 floats (4x b128) each
    const int ar = tid >> 2;
    const int ac = (tid & 3) * 16;

    const unsigned asbase = (unsigned)(uintptr_t)&Asf[0][0][0];
    const unsigned bsbase = (unsigned)(uintptr_t)&Bsf[0][0][0];

    // fragment gather indices (CDNA5 16-bit WMMA layouts)
    const int arow = mo + (lane & 15);
    const int kb   = (lane >> 4) * 8;   // lanes 0-15: K 0..7/16..23 ; lanes 16-31: 8..15/24..31

    // issue one tile pair (A,B) of async global->LDS copies: 8 b128 per thread
    auto issue_tile = [&](int k0, int buf) {
        unsigned aoff = asbase + (unsigned)(buf * ABUF + ar * (BK + AP) + ac) * 4u;
        unsigned boff = bsbase + (unsigned)(buf * BBUF + ar * (BN + BP) + ac) * 4u;
        unsigned aga = ((unsigned)(m0 + ar) * (unsigned)K + (unsigned)(k0 + ac)) * 4u;
        unsigned bga = ((unsigned)(k0 + ar) * (unsigned)N + (unsigned)(n0 + ac)) * 4u;
        asm volatile(
            "global_load_async_to_lds_b128 %0, %4, %5 offset:0\n\t"
            "global_load_async_to_lds_b128 %1, %4, %5 offset:16\n\t"
            "global_load_async_to_lds_b128 %2, %4, %5 offset:32\n\t"
            "global_load_async_to_lds_b128 %3, %4, %5 offset:48"
            :
            : "v"(aoff), "v"(aoff + 16u), "v"(aoff + 32u), "v"(aoff + 48u),
              "v"(aga), "s"(A)
            : "memory");
        asm volatile(
            "global_load_async_to_lds_b128 %0, %4, %5 offset:0\n\t"
            "global_load_async_to_lds_b128 %1, %4, %5 offset:16\n\t"
            "global_load_async_to_lds_b128 %2, %4, %5 offset:32\n\t"
            "global_load_async_to_lds_b128 %3, %4, %5 offset:48"
            :
            : "v"(boff), "v"(boff + 16u), "v"(boff + 32u), "v"(boff + 48u),
              "v"(bga), "s"(B)
            : "memory");
    };

    const int nk = K / BK;
    issue_tile(0, 0);                               // prologue: tile 0 in flight

    for (int kt = 0; kt < nk; ++kt) {
        const int buf = kt & 1;
        if (kt + 1 < nk) {
            issue_tile((kt + 1) * BK, buf ^ 1);     // prefetch next tile (16 in flight)
            asm volatile("s_wait_asynccnt 8" ::: "memory");   // tile kt landed (in-order)
        } else {
            asm volatile("s_wait_asynccnt 0" ::: "memory");
        }
        __syncthreads();                            // all threads' tile-kt data visible

        // ---- two 16x16x32 K-steps per LDS buffer ----
        #pragma unroll
        for (int s = 0; s < 2; ++s) {
            const int ks = s * 32;
            v16bf af, bf0, bf1;
            #pragma unroll
            for (int i = 0; i < 8; ++i) {
                af[i]     = (__bf16)Asf[buf][arow][ks + kb + i];
                af[i + 8] = (__bf16)Asf[buf][arow][ks + kb + 16 + i];
            }
            #pragma unroll
            for (int i = 0; i < 8; ++i) {
                bf0[i]     = (__bf16)Bsf[buf][ks + kb + i][no + (lane & 15)];   // lane%16 = N, elements = K (same order as A)
                bf0[i + 8] = (__bf16)Bsf[buf][ks + kb + 16 + i][no + (lane & 15)];
                bf1[i]     = (__bf16)Bsf[buf][ks + kb + i][no + 16 + (lane & 15)];
                bf1[i + 8] = (__bf16)Bsf[buf][ks + kb + 16 + i][no + 16 + (lane & 15)];
            }
            acc0 = __builtin_amdgcn_wmma_f32_16x16x32_bf16(false, af, false, bf0,
                                                           (short)0, acc0, false, false);
            acc1 = __builtin_amdgcn_wmma_f32_16x16x32_bf16(false, af, false, bf1,
                                                           (short)0, acc1, false, false);
        }
        __syncthreads();                 // reads retired before this buffer is refilled
    }

    // C/D layout: element r -> row r (lanes 0-15) or row 8+r (lanes 16-31), col = lane%16
    const int rowb = m0 + mo + (lane >> 4) * 8;
    const int col0 = n0 + no + (lane & 15);
    #pragma unroll
    for (int r = 0; r < 8; ++r) {
        int row = rowb + r;
        float v0 = acc0[r], v1 = acc1[r];
        if (rowscale) { float s = rowscale[(size_t)row * rs_stride]; v0 *= s; v1 *= s; }
        if (bias)     { v0 += bias[col0]; v1 += bias[col0 + 16]; }
        float* c0 = C + (size_t)row * N + col0;
        float* c1 = c0 + 16;
        if (accum) { *c0 += v0; *c1 += v1; } else { *c0 = v0; *c1 = v1; }
    }
}

// ---------------- RoPE (interleaved pairs) ----------------
__global__ void rope_kernel(float* __restrict__ x, const int* __restrict__ start, int nheads) {
    int idx = blockIdx.x * blockDim.x + threadIdx.x;
    int total = T_ * nheads * (DH_ / 2);
    if (idx >= total) return;
    int i = idx % (DH_ / 2);
    int h = (idx / (DH_ / 2)) % nheads;
    int t = idx / ((DH_ / 2) * nheads);
    float pos  = (float)(start[0] + t);
    float freq = __expf(-((float)(2 * i) / (float)DH_) * 9.210340371976184f); // ln(1e4)
    float ang  = pos * freq;
    float s, c;
    __sincosf(ang, &s, &c);
    float* p = x + ((size_t)t * nheads + h) * DH_ + 2 * i;
    float xr = p[0], xi = p[1];
    p[0] = xr * c - xi * s;
    p[1] = xr * s + xi * c;
}

// ---------------- sliding-window GQA attention (one block per (t, head)) ----------------
__global__ __launch_bounds__(128)
void attn_kernel(const float* __restrict__ q, const float* __restrict__ k,
                 const float* __restrict__ v, float* __restrict__ o) {
    const int t   = blockIdx.x;
    const int hh  = blockIdx.y;
    const int kvh = hh / (H_ / KV_);
    const int tid = threadIdx.x;

    __shared__ float qs[DH_];
    __shared__ float sc[WINDOW_];
    __shared__ float red[128];

    const float* qrow = q + ((size_t)t * H_ + hh) * DH_;
    if (tid < DH_) qs[tid] = qrow[tid];
    __syncthreads();

    int kstart = t - (WINDOW_ - 1); if (kstart < 0) kstart = 0;
    int cnt = t - kstart + 1;

    float lmax = -1e30f;
    for (int kk = tid; kk < cnt; kk += 128) {
        const float* krow = k + ((size_t)(kstart + kk) * KV_ + kvh) * DH_;
        float dot = 0.f;
        #pragma unroll 8
        for (int d = 0; d < DH_; ++d) dot += qs[d] * krow[d];
        dot *= 0.125f;                     // 1/sqrt(64)
        sc[kk] = dot;
        lmax = fmaxf(lmax, dot);
    }
    red[tid] = lmax; __syncthreads();
    for (int s = 64; s > 0; s >>= 1) {
        if (tid < s) red[tid] = fmaxf(red[tid], red[tid + s]);
        __syncthreads();
    }
    float mx = red[0];
    __syncthreads();

    float lsum = 0.f;
    for (int kk = tid; kk < cnt; kk += 128) {
        float p = __expf(sc[kk] - mx);
        sc[kk] = p;
        lsum += p;
    }
    red[tid] = lsum; __syncthreads();
    for (int s = 64; s > 0; s >>= 1) {
        if (tid < s) red[tid] += red[tid + s];
        __syncthreads();
    }
    float inv = 1.0f / red[0];

    for (int d = tid; d < DH_; d += 128) {
        float acc = 0.f;
        for (int kk = 0; kk < cnt; ++kk)
            acc += sc[kk] * v[((size_t)(kstart + kk) * KV_ + kvh) * DH_ + d];
        o[((size_t)t * H_ + hh) * DH_ + d] = acc * inv;
    }
}

// ---------------- router: logits, top-2, softmax weights ----------------
__global__ void gate_kernel(const float* __restrict__ xf, const float* __restrict__ gw,
                            float* __restrict__ full) {
    int t = blockIdx.x * blockDim.x + threadIdx.x;
    if (t >= T_) return;
    float lg[E_];
    #pragma unroll
    for (int e = 0; e < E_; ++e) lg[e] = 0.f;
    const float* xr = xf + (size_t)t * D_;
    for (int d = 0; d < D_; ++d) {
        float xv = xr[d];
        const float* g = gw + (size_t)d * E_;
        #pragma unroll
        for (int e = 0; e < E_; ++e) lg[e] += xv * g[e];
    }
    int i0 = 0;
    for (int e = 1; e < E_; ++e) if (lg[e] > lg[i0]) i0 = e;
    int i1 = (i0 == 0) ? 1 : 0;
    for (int e = 0; e < E_; ++e) { if (e == i0) continue; if (lg[e] > lg[i1]) i1 = e; }
    float e1 = __expf(lg[i1] - lg[i0]);
    float s  = 1.0f + e1;
    float* fr = full + (size_t)t * E_;
    #pragma unroll
    for (int e = 0; e < E_; ++e) fr[e] = 0.f;
    fr[i0] = 1.0f / s;
    fr[i1] = e1 / s;
}

// ---------------- SwiGLU elementwise: t1 = silu(t1) * t2 ----------------
__global__ void silu_mul_kernel(float* __restrict__ t1, const float* __restrict__ t2, int n) {
    int i = blockIdx.x * blockDim.x + threadIdx.x;
    if (i >= n) return;
    float a = t1[i];
    t1[i] = a * (1.0f / (1.0f + __expf(-a))) * t2[i];
}

// ---------------- host-side launcher ----------------
static inline void launch_gemm(const float* A, const float* Bw, float* C,
                               const float* bias, const float* rowscale, int rs_stride,
                               int M, int N, int K, int accum, hipStream_t s) {
    dim3 grid(N / 64, M / 64);
    gemm_bf16_wmma<<<grid, 256, 0, s>>>(A, Bw, C, bias, rowscale, rs_stride, M, N, K, accum);
}

extern "C" void kernel_launch(void* const* d_in, const int* in_sizes, int n_in,
                              void* d_out, int out_size, void* d_ws, size_t ws_size,
                              hipStream_t stream) {
    (void)in_sizes; (void)n_in; (void)out_size; (void)ws_size;
    const int*   x            = (const int*)  d_in[0];
    const int*   start_pos    = (const int*)  d_in[1];
    const float* emb          = (const float*)d_in[2];
    const float* attn_norm_w  = (const float*)d_in[3];
    const float* wq           = (const float*)d_in[4];
    const float* wk           = (const float*)d_in[5];
    const float* wv           = (const float*)d_in[6];
    const float* wo           = (const float*)d_in[7];
    const float* ffn_norm_w   = (const float*)d_in[8];
    const float* gate_w       = (const float*)d_in[9];
    const float* w1           = (const float*)d_in[10];
    const float* w3           = (const float*)d_in[11];
    const float* w2           = (const float*)d_in[12];
    const float* final_norm_w = (const float*)d_in[13];
    const float* out_w        = (const float*)d_in[14];
    const float* out_b        = (const float*)d_in[15];
    float* out = (float*)d_out;

    float* ws = (float*)d_ws;
    size_t off = 0;
    float* h   = ws + off; off += (size_t)T_ * D_;
    float* xa  = ws + off; off += (size_t)T_ * D_;
    float* q   = ws + off; off += (size_t)T_ * H_ * DH_;
    float* kb  = ws + off; off += (size_t)T_ * KV_ * DH_;
    float* vb  = ws + off; off += (size_t)T_ * KV_ * DH_;
    float* o   = ws + off; off += (size_t)T_ * D_;
    float* xf  = ws + off; off += (size_t)T_ * D_;
    float* t1  = ws + off; off += (size_t)T_ * F_;
    float* t2  = ws + off; off += (size_t)T_ * F_;
    float* fullw = ws + off; off += (size_t)T_ * E_;

    embed_kernel<<<(T_ * D_ + 255) / 256, 256, 0, stream>>>(x, emb, h);

    for (int l = 0; l < L_; ++l) {
        // ---- attention block ----
        rmsnorm_kernel<<<T_, 256, 0, stream>>>(h, attn_norm_w + (size_t)l * D_, xa);
        launch_gemm(xa, wq + (size_t)l * D_ * (H_ * DH_),  q,  nullptr, nullptr, 0,
                    T_, H_ * DH_,  D_, 0, stream);
        launch_gemm(xa, wk + (size_t)l * D_ * (KV_ * DH_), kb, nullptr, nullptr, 0,
                    T_, KV_ * DH_, D_, 0, stream);
        launch_gemm(xa, wv + (size_t)l * D_ * (KV_ * DH_), vb, nullptr, nullptr, 0,
                    T_, KV_ * DH_, D_, 0, stream);
        rope_kernel<<<(T_ * H_ * (DH_ / 2) + 255) / 256, 256, 0, stream>>>(q,  start_pos, H_);
        rope_kernel<<<(T_ * KV_ * (DH_ / 2) + 255) / 256, 256, 0, stream>>>(kb, start_pos, KV_);
        dim3 agrid(S_, H_);
        attn_kernel<<<agrid, 128, 0, stream>>>(q, kb, vb, o);
        // h += o @ wo
        launch_gemm(o, wo + (size_t)l * (H_ * DH_) * D_, h, nullptr, nullptr, 0,
                    T_, D_, H_ * DH_, 1, stream);

        // ---- MoE block ----
        rmsnorm_kernel<<<T_, 256, 0, stream>>>(h, ffn_norm_w + (size_t)l * D_, xf);
        gate_kernel<<<(T_ + 255) / 256, 256, 0, stream>>>(xf, gate_w + (size_t)l * D_ * E_, fullw);
        for (int e = 0; e < E_; ++e) {
            const float* w1e = w1 + ((size_t)l * E_ + e) * D_ * F_;
            const float* w3e = w3 + ((size_t)l * E_ + e) * D_ * F_;
            const float* w2e = w2 + ((size_t)l * E_ + e) * F_ * D_;
            launch_gemm(xf, w1e, t1, nullptr, nullptr, 0, T_, F_, D_, 0, stream);
            launch_gemm(xf, w3e, t2, nullptr, nullptr, 0, T_, F_, D_, 0, stream);
            silu_mul_kernel<<<(T_ * F_ + 255) / 256, 256, 0, stream>>>(t1, t2, T_ * F_);
            // h += full[:, e] * (t1 @ w2e)   (rows with zero gate weight add zero)
            launch_gemm(t1, w2e, h, nullptr, fullw + e, E_, T_, D_, F_, 1, stream);
        }
    }

    // ---- final norm + vocab projection ----
    rmsnorm_kernel<<<T_, 256, 0, stream>>>(h, final_norm_w, xa);
    launch_gemm(xa, out_w, out, out_b, nullptr, 0, T_, V_, D_, 0, stream);
}
